// Generator_12635793785137
// MI455X (gfx1250) — matrix operand order repaired and checked by reading, not compiled, not run on hardware
//
#include <hip/hip_runtime.h>
#include <math.h>

typedef __attribute__((ext_vector_type(16))) __bf16 v16bf;
typedef __attribute__((ext_vector_type(8)))  float  v8f;
typedef __attribute__((ext_vector_type(4)))  float  f4;   // trivial union member (HIP float4 has ctors)
typedef unsigned short bfu;

#define B_  256
#define N_  32
#define H_  256
#define T_  496

union BfVec { f4 f2[2]; v16bf v; };

__device__ __forceinline__ bfu f2bf(float f) {
  unsigned int x = __float_as_uint(f);
  x += 0x7FFFu + ((x >> 16) & 1u);           // round-to-nearest-even
  return (bfu)(x >> 16);
}
__device__ __forceinline__ float sigf(float x) { return 1.f / (1.f + __expf(-x)); }

__device__ __forceinline__ v8f wmma_bf16(v16bf a, v16bf b, v8f c) {
  return __builtin_amdgcn_wmma_f32_16x16x32_bf16(false, a, false, b, (short)0, c, false, false);
}

// ---- opaque weight-fragment loads: volatile asm so LICM cannot hoist/spill them ----
// 8 B fragments (16 x global_load_b128) from ONE base address + constant byte offset;
// base already includes wave (w*8192B) and lane (lane*32B) components, so only a single
// 64-bit VGPR address pair stays live for the whole GEMM (no address spills).
__device__ __forceinline__ void issue_B8(const bfu* base, int byteoff, BfVec* d) {
#pragma unroll
  for (int kt = 0; kt < 8; ++kt) {
    asm volatile("global_load_b128 %0, %2, off offset:%3\n\t"
                 "global_load_b128 %1, %2, off offset:%4"
                 : "=v"(d[kt].f2[0]), "=v"(d[kt].f2[1])
                 : "v"(base), "i"(byteoff + kt * 1024), "i"(byteoff + kt * 1024 + 16));
  }
}
// wait until only the newest 16 loads (next buffer) are outstanding / until all done
#define WAIT_LD16() asm volatile("s_wait_loadcnt 0x10" ::: "memory")
#define WAIT_LD0()  asm volatile("s_wait_loadcnt 0x0"  ::: "memory")
// pin buffer register definitions after the wait so WMMAs cannot float above it
__device__ __forceinline__ void tie8(BfVec* d) {
#pragma unroll
  for (int kt = 0; kt < 8; ++kt)
    asm volatile("" : "+v"(d[kt].f2[0]), "+v"(d[kt].f2[1]));
}

// ---------------- setup: pack weights into WMMA B-fragment layout (bf16) ----------------
// block (nt,kt): 32 lanes x 16 bf16; lane L: N = nt*16 + (L&15), K = kt*32 + (L>>4)*8 + (e&7) + (e>>3)*16
__global__ void pack_kernel(const float* __restrict__ Wihn, const float* __restrict__ Whhn,
                            const float* __restrict__ Wihg, const float* __restrict__ Whhg,
                            const float* __restrict__ Ws1, bfu* __restrict__ packed) {
  int e = blockIdx.x * blockDim.x + threadIdx.x;
  if (e >= 802816) return;
  const int R0 = 393216, R1 = 786432;
  int region, le;
  if (e < R0)      { region = 0; le = e; }
  else if (e < R1) { region = 1; le = e - R0; }
  else             { region = 2; le = e - R1; }
  int blk = le >> 9, within = le & 511;
  int lane = within >> 4, e16 = within & 15;
  int nt = blk >> 3, kt = blk & 7;
  int K = kt * 32 + (lane >> 4) * 8 + (e16 & 7) + (e16 >> 3) * 16;
  int n = nt * 16 + (lane & 15);
  float val;
  if (region == 0)       val = (n < 768) ? Wihn[n * 270 + 14 + K] : Whhn[(n - 768) * 256 + K];
  else if (region == 1)  val = (n < 768) ? Wihg[n * 256 + K]      : Whhg[(n - 768) * 256 + K];
  else                   val = Ws1[n * 256 + K];
  packed[e] = f2bf(val);
}

// colT[c][n] = Wih_node[n][c] for c in 0..13 (one-hot column gathers); idx = argmax(one-hot)
__global__ void setup2_kernel(const float* __restrict__ Wihn, const float* __restrict__ nf,
                              float* __restrict__ colT, int* __restrict__ idxg) {
  int e = blockIdx.x * blockDim.x + threadIdx.x;
  if (e < 10752) {
    int c = e / 768, n = e - c * 768;
    colT[e] = Wihn[n * 270 + c];
  } else if (e < 10752 + 8192) {
    int q = e - 10752;
    const float* p = nf + q * 7;
    int best = 0;
    for (int f = 1; f < 7; ++f) if (p[f] > p[best]) best = f;
    idxg[q] = best;
  }
}

// ---------------- main recurrence: 16 WGs x 16 batch rows, 8 waves each ----------------
__global__ __launch_bounds__(256) void recur_kernel(
    const float* __restrict__ z, const int* __restrict__ idxg, const float* __restrict__ colT,
    const bfu* __restrict__ pWn, const bfu* __restrict__ pWg, const bfu* __restrict__ pW1,
    const float* __restrict__ bih, const float* __restrict__ bhh,
    const float* __restrict__ bihg, const float* __restrict__ bhhg,
    const float* __restrict__ bs1, const float* __restrict__ Ws2,
    const float* __restrict__ bs2, float* __restrict__ probs) {
  __shared__ __align__(16) float hNf[16 * H_];
  __shared__ __align__(16) float hGf[16 * H_];
  __shared__ __align__(16) bfu   hNb[16 * H_];
  __shared__ __align__(16) bfu   hGb[16 * H_];
  __shared__ __align__(16) float hidl[16 * 64];
  __shared__ int ci_l[16];
  __shared__ int cj_l[16];

  const int tid  = threadIdx.x;
  const int w    = tid >> 5;
  const int lane = tid & 31;
  const int l15  = lane & 15;
  const int kh   = lane >> 4;
  const int m0   = blockIdx.x * 16;

  // single live base address per packed matrix: wave + lane components folded in
  const bfu* bpn = pWn + (w << 12) + lane * 16;   // nt = w + 8q -> imm off q*65536
  const bfu* bpg = pWg + (w << 12) + lane * 16;
  const bfu* bp1 = pW1 + (w << 12) + lane * 16;

  for (int e = tid; e < 16 * H_; e += 256) {
    int m = e >> 8, k = e & 255;
    float v = z[(m0 + m) * H_ + k];
    hNf[e] = v; hGf[e] = v;
    bfu bv = f2bf(v);
    hNb[e] = bv; hGb[e] = bv;
  }
  __syncthreads();

  for (int i = 0; i < N_ - 1; ++i) {
    if (tid < 16) ci_l[tid] = idxg[(m0 + tid) * N_ + i];
    __syncthreads();

    for (int j = i + 1; j < N_; ++j) {
      if (tid < 16) cj_l[tid] = idxg[(m0 + tid) * N_ + j];
      __syncthreads();

      // A fragments: node_h bf16 from LDS (documented 16-bit A 16x32 layout)
      v16bf A[8];
#pragma unroll
      for (int kt = 0; kt < 8; ++kt) {
        int k0 = kt * 32 + kh * 8;
        BfVec bv;
        bv.f2[0] = *(const f4*)&hNb[l15 * H_ + k0];
        bv.f2[1] = *(const f4*)&hNb[l15 * H_ + k0 + 16];
        A[kt] = bv.v;
      }
      __syncthreads();

      // fused gate GEMM: [16x256] x [256x1536] (gi_h | gh), wave w owns ntiles nt%8==w
      // double-buffered L2 weight streaming, software pipelined against WMMA
      v8f acc[12];
#pragma unroll
      for (int q = 0; q < 12; ++q)
#pragma unroll
        for (int e = 0; e < 8; ++e) acc[q][e] = 0.f;

      BfVec buf[2][8];
      issue_B8(bpn, 0, buf[0]);
#pragma unroll
      for (int q = 0; q < 12; ++q) {
        if (q < 11) {
          issue_B8(bpn, (q + 1) * 65536, buf[(q + 1) & 1]);
          WAIT_LD16();
        } else {
          WAIT_LD0();
        }
        tie8(buf[q & 1]);
#pragma unroll
        for (int kt = 0; kt < 8; ++kt)
          acc[q] = wmma_bf16(A[kt], buf[q & 1][kt].v, acc[q]);
      }

      // GRU nonlinearity fully in registers (gate triples co-owned by construction)
#pragma unroll
      for (int qt = 0; qt < 2; ++qt) {
        int t  = w + 8 * qt;
        int hc = t * 16 + l15;
        float bir  = bih[hc],       bhr = bhh[hc];
        float biz  = bih[256 + hc], bhz = bhh[256 + hc];
        float bin_ = bih[512 + hc], bhn = bhh[512 + hc];
#pragma unroll
        for (int v = 0; v < 8; ++v) {
          int m = v + 8 * kh;
          int ci = ci_l[m], cj = cj_l[m] + 7;
          float gir = acc[qt][v]      + colT[ci * 768 + hc]       + colT[cj * 768 + hc]       + bir;
          float giz = acc[qt + 2][v]  + colT[ci * 768 + 256 + hc] + colT[cj * 768 + 256 + hc] + biz;
          float gin = acc[qt + 4][v]  + colT[ci * 768 + 512 + hc] + colT[cj * 768 + 512 + hc] + bin_;
          float ghr = acc[qt + 6][v]  + bhr;
          float ghz = acc[qt + 8][v]  + bhz;
          float ghn = acc[qt + 10][v] + bhn;
          float r  = sigf(gir + ghr);
          float zg = sigf(giz + ghz);
          float ng = tanhf(gin + r * ghn);
          float ho = hNf[m * H_ + hc];
          float hn = (1.f - zg) * ng + zg * ho;
          hNf[m * H_ + hc] = hn;
          hNb[m * H_ + hc] = f2bf(hn);
        }
      }
      __syncthreads();

      // edge head: hid = relu(h' @ Ws1^T + bs1), 4 N-tiles on waves 0..3
      if (w < 4) {
        BfVec hb[8];
        issue_B8(bp1, 0, hb);
        v16bf A2[8];
#pragma unroll
        for (int kt = 0; kt < 8; ++kt) {
          int k0 = kt * 32 + kh * 8;
          BfVec bv;
          bv.f2[0] = *(const f4*)&hNb[l15 * H_ + k0];
          bv.f2[1] = *(const f4*)&hNb[l15 * H_ + k0 + 16];
          A2[kt] = bv.v;
        }
        WAIT_LD0();
        tie8(hb);
        v8f a2;
#pragma unroll
        for (int e = 0; e < 8; ++e) a2[e] = 0.f;
#pragma unroll
        for (int kt = 0; kt < 8; ++kt)
          a2 = wmma_bf16(A2[kt], hb[kt].v, a2);
        int n = w * 16 + l15;
        float b1 = bs1[n];
#pragma unroll
        for (int v = 0; v < 8; ++v) {
          int m = v + 8 * kh;
          float hv = a2[v] + b1;
          hidl[m * 64 + n] = hv > 0.f ? hv : 0.f;
        }
      }
      __syncthreads();

      if (w == 7 && lane < 16) {
        int m = lane;
        float s = bs2[0];
        for (int c = 0; c < 64; ++c) s += hidl[m * 64 + c] * Ws2[c];
        probs[(i * N_ + j) * B_ + m0 + m] = sigf(s);
      }
      __syncthreads();
    }

    // graph-level GRU: graph_h' = GRU(x=node_h, h=graph_h); node_h := graph_h'
    if (i != N_ - 2) {
      v16bf An[8], Ag[8];
#pragma unroll
      for (int kt = 0; kt < 8; ++kt) {
        int k0 = kt * 32 + kh * 8;
        BfVec bn, bg;
        bn.f2[0] = *(const f4*)&hNb[l15 * H_ + k0];
        bn.f2[1] = *(const f4*)&hNb[l15 * H_ + k0 + 16];
        bg.f2[0] = *(const f4*)&hGb[l15 * H_ + k0];
        bg.f2[1] = *(const f4*)&hGb[l15 * H_ + k0 + 16];
        An[kt] = bn.v; Ag[kt] = bg.v;
      }
      __syncthreads();

      v8f acc[12];
#pragma unroll
      for (int q = 0; q < 12; ++q)
#pragma unroll
        for (int e = 0; e < 8; ++e) acc[q][e] = 0.f;

      BfVec buf[2][8];
      issue_B8(bpg, 0, buf[0]);
#pragma unroll
      for (int q = 0; q < 12; ++q) {
        if (q < 11) {
          issue_B8(bpg, (q + 1) * 65536, buf[(q + 1) & 1]);
          WAIT_LD16();
        } else {
          WAIT_LD0();
        }
        tie8(buf[q & 1]);
#pragma unroll
        for (int kt = 0; kt < 8; ++kt)
          acc[q] = wmma_bf16((q < 6) ? An[kt] : Ag[kt], buf[q & 1][kt].v, acc[q]);
      }

#pragma unroll
      for (int qt = 0; qt < 2; ++qt) {
        int t  = w + 8 * qt;
        int hc = t * 16 + l15;
        float bir  = bihg[hc],       bhr = bhhg[hc];
        float biz  = bihg[256 + hc], bhz = bhhg[256 + hc];
        float bin_ = bihg[512 + hc], bhn = bhhg[512 + hc];
#pragma unroll
        for (int v = 0; v < 8; ++v) {
          int m = v + 8 * kh;
          float r  = sigf(acc[qt][v] + bir + acc[qt + 6][v] + bhr);
          float zg = sigf(acc[qt + 2][v] + biz + acc[qt + 8][v] + bhz);
          float ng = tanhf(acc[qt + 4][v] + bin_ + r * (acc[qt + 10][v] + bhn));
          float ho = hGf[m * H_ + hc];
          float hn = (1.f - zg) * ng + zg * ho;
          bfu hb = f2bf(hn);
          hGf[m * H_ + hc] = hn; hGb[m * H_ + hc] = hb;
          hNf[m * H_ + hc] = hn; hNb[m * H_ + hc] = hb;
        }
      }
      __syncthreads();
    }
  }
}

// ---------------- epilogue: CB-rsample + anti-diagonal adjacency scatter ----------------
__global__ void adj_kernel(const float* __restrict__ probs, const float* __restrict__ u,
                           float* __restrict__ out) {
  int g = blockIdx.x * blockDim.x + threadIdx.x;
  if (g >= B_ * T_) return;
  int b = g / T_, t = g - b * T_;
  int rem = t, i = 0;
  while (rem >= N_ - 1 - i) { rem -= N_ - 1 - i; ++i; }   // row-major triu pair
  int j = i + 1 + rem;
  float p0 = probs[(i * N_ + j) * B_ + b];
  float uu = u[b * T_ + t];
  const float EPS = 1.1920929e-07f;
  float p = fminf(fmaxf(p0, EPS), 1.f - EPS);
  bool outside = (p < 0.499f) || (p > 0.501f);
  float cut = outside ? p : 0.499f;
  float num = log1pf(-cut + uu * (2.f * cut - 1.f)) - log1pf(-cut);
  float den = logf(cut) - log1pf(-cut);
  float val = outside ? (num / den) : uu;
  int k = t, d = 1;
  while (k >= d) { k -= d; ++d; }                          // anti-diagonal placement
  int r = k, c = k + N_ - d;
  float* adj = out + B_ * N_ * 7;
  adj[b * 1024 + r * N_ + c] = val;
  adj[b * 1024 + c * N_ + r] = val;
}

__global__ void misc_kernel(const float* __restrict__ nf, float* __restrict__ out) {
  int g = blockIdx.x * blockDim.x + threadIdx.x;
  if (g < 57344) {
    out[g] = nf[g];                                        // x = node_features flat
  } else if (g < 65536) {
    int q = g - 57344;
    ((int*)(out + 57344 + 262144))[q] = q >> 5;            // batch (int32)
  } else if (g < 73728) {
    int q = g - 65536;
    int b = q >> 5, dd = q & 31;
    out[57344 + b * 1024 + dd * 33] = 0.f;                 // zero adjacency diagonal
  }
}

extern "C" void kernel_launch(void* const* d_in, const int* in_sizes, int n_in,
                              void* d_out, int out_size, void* d_ws, size_t ws_size,
                              hipStream_t stream) {
  (void)in_sizes; (void)n_in; (void)out_size; (void)ws_size;
  const float* z    = (const float*)d_in[0];
  const float* nf   = (const float*)d_in[1];
  const float* u    = (const float*)d_in[2];
  const float* Wihn = (const float*)d_in[3];
  const float* Whhn = (const float*)d_in[4];
  const float* bih  = (const float*)d_in[5];
  const float* bhh  = (const float*)d_in[6];
  const float* Wihg = (const float*)d_in[7];
  const float* Whhg = (const float*)d_in[8];
  const float* bihg = (const float*)d_in[9];
  const float* bhhg = (const float*)d_in[10];
  const float* Ws1  = (const float*)d_in[11];
  const float* bs1  = (const float*)d_in[12];
  const float* Ws2  = (const float*)d_in[13];
  const float* bs2  = (const float*)d_in[14];
  float* out = (float*)d_out;

  char* ws = (char*)d_ws;                 // total workspace use: 2,697,216 bytes
  bfu*   pWn  = (bfu*)(ws);               // 786432 B packed [WihH|Whh] node
  bfu*   pWg  = (bfu*)(ws + 786432);      // 786432 B packed [Wih|Whh] graph
  bfu*   pW1  = (bfu*)(ws + 1572864);     //  32768 B packed Ws1
  float* colT = (float*)(ws + 1605632);   //  43008 B one-hot column table
  int*   idxg = (int*)(ws + 1648640);     //  32768 B node class indices
  float* prob = (float*)(ws + 1681408);   // 1015808 B prob grid [31][32][256]

  pack_kernel<<<(802816 + 255) / 256, 256, 0, stream>>>(Wihn, Whhn, Wihg, Whhg, Ws1, (bfu*)ws);
  setup2_kernel<<<(10752 + 8192 + 255) / 256, 256, 0, stream>>>(Wihn, nf, colT, idxg);
  recur_kernel<<<16, 256, 0, stream>>>(z, idxg, colT, pWn, pWg, pW1,
                                       bih, bhh, bihg, bhhg, bs1, Ws2, bs2, prob);
  adj_kernel<<<(B_ * T_ + 255) / 256, 256, 0, stream>>>(prob, u, out);
  misc_kernel<<<73728 / 256, 256, 0, stream>>>(nf, out);
}
